// MultiHeadAttention_1194000908544
// MI455X (gfx1250) — matrix operand order repaired and checked by reading, not compiled, not run on hardware
//
#include <hip/hip_runtime.h>

// ---------------------------------------------------------------- constants
constexpr int CB  = 4;
constexpr int CT  = 2048;
constexpr int CH  = 16;
constexpr int CD  = 1024;
constexpr int CHD = 64;
constexpr int CBT = CB * CT;          // 8192 rows total

// ---------------------------------------------------------------- types
typedef __attribute__((ext_vector_type(16))) __bf16 v16bf;
typedef __attribute__((ext_vector_type(8)))  __bf16 v8bf;
typedef __attribute__((ext_vector_type(8)))  float  v8f;
typedef __attribute__((ext_vector_type(4)))  unsigned int u32x4;
typedef __attribute__((ext_vector_type(8)))  unsigned int u32x8;

// Build a 16-element bf16 fragment from two 16B halves (p0 -> elems 0..7,
// p1 -> elems 8..15). Used for A fragments (halves at +0/+16 elems).
__device__ inline v16bf ld_frag(const __bf16* p0, const __bf16* p1) {
  u32x4 a = *(const u32x4*)p0;
  u32x4 b = *(const u32x4*)p1;
  u32x8 t;
  t[0] = a[0]; t[1] = a[1]; t[2] = a[2]; t[3] = a[3];
  t[4] = b[0]; t[5] = b[1]; t[6] = b[2]; t[7] = b[3];
  return __builtin_bit_cast(v16bf, t);
}

__device__ inline v16bf ld_colk(const __bf16* p, int stride, int hi) {
  v16bf r;
#pragma unroll
  for (int s = 0; s < 16; ++s) {
    int k = (s < 8) ? (8 * hi + s) : (16 + 8 * hi + (s - 8));
    r[s] = p[k * stride];
  }
  return r;
}

__device__ inline v8f wmma_bf16(v16bf a, v16bf b, v8f c) {
  // D = A(16x32 bf16) * B(32x16 bf16) + C(16x16 f32)
  return __builtin_amdgcn_wmma_f32_16x16x32_bf16(
      false, a, false, b, (short)0, c, false, false);
}

// ------------------------------------------------ CDNA5 async LDS staging
// Generic pointers to LDS are {shared_aperture, ds_offset}; the low 32 bits
// are the DS byte offset the async-load VDST / TDM lds_addr fields expect.
__device__ inline unsigned lds_off(const void* p) {
  return (unsigned)(unsigned long long)(uintptr_t)p;
}

// GLOBAL_LOAD_ASYNC_TO_LDS_B128: 16B memory -> LDS per lane, ASYNCcnt-tracked.
__device__ inline void async_ld_b128(unsigned lds, const void* g) {
  unsigned long long ga = (unsigned long long)(uintptr_t)g;
  asm volatile("global_load_async_to_lds_b128 %0, %1, off"
               :: "v"(lds), "v"(ga) : "memory");
}

__device__ inline void wait_async() {
#if __has_builtin(__builtin_amdgcn_s_wait_asynccnt)
  __builtin_amdgcn_s_wait_asynccnt(0);
#else
  asm volatile("s_wait_asynccnt 0x0" ::: "memory");
#endif
}

// ------------------------------------------------ CDNA5 Tensor Data Mover
// 2D tile DMA: global (row-major, row_stride elems of 2B) -> LDS (packed
// tile_w x tile_h), one TENSOR_LOAD_TO_LDS with a hand-built D# (groups 0+1,
// VADDR2/3 = NULL -> <=2D form). TENSORcnt-tracked.
__device__ inline void tdm_load_tile_2d(unsigned lds_addr, const void* gptr,
                                        unsigned tile_w, unsigned tile_h,
                                        unsigned row_stride) {
  unsigned long long ga = (unsigned long long)(uintptr_t)gptr;
  u32x4 g0;
  g0[0] = 1u;                                              // count=1 (valid, user)
  g0[1] = lds_addr;                                        // lds_addr[63:32]
  g0[2] = (unsigned)ga;                                    // global_addr lo
  g0[3] = (unsigned)((ga >> 32) & 0x01ffffffu) | (2u << 30); // addr hi | type=2
  // group1 bit layout (ISA 8.4), data_size=1 (2 bytes):
  unsigned long long w0 = (1ull << 16)                                   // data_size
                        | ((unsigned long long)(tile_w & 0xffffu) << 48); // tensor_dim0 lo16
  unsigned long long w1 = ((unsigned long long)tile_h << 16)              // tensor_dim1
                        | ((unsigned long long)(tile_w & 0xffffu) << 48); // tile_dim0
  unsigned long long w2 = (unsigned long long)(tile_h & 0xffffu)          // tile_dim1
                        | ((unsigned long long)row_stride << 32);         // dim0_stride lo32
  unsigned long long w3 = 0ull;                                           // stride hi, dim1_stride
  u32x8 g1;
  g1[0] = (unsigned)w0; g1[1] = (unsigned)(w0 >> 32);
  g1[2] = (unsigned)w1; g1[3] = (unsigned)(w1 >> 32);
  g1[4] = (unsigned)w2; g1[5] = (unsigned)(w2 >> 32);
  g1[6] = (unsigned)w3; g1[7] = (unsigned)(w3 >> 32);
  asm volatile("tensor_load_to_lds %0, %1" :: "s"(g0), "s"(g1) : "memory");
}

__device__ inline void wait_tensor() {
  __builtin_amdgcn_s_wait_tensorcnt(0);
}

// ---------------------------------------------------------------- converts
__global__ void cvt_bf16(const float* __restrict__ src, __bf16* __restrict__ dst, int n) {
  int i = blockIdx.x * blockDim.x + threadIdx.x;
  if (i < n) dst[i] = (__bf16)src[i];
}

// dst[c*rows + r] = src[r*cols + c]   (Wo[j][d] -> WoT[d][j])
__global__ void cvt_bf16_T(const float* __restrict__ src, __bf16* __restrict__ dst,
                           int rows, int cols) {
  int i = blockIdx.x * blockDim.x + threadIdx.x;
  if (i < rows * cols) {
    int r = i / cols, c = i % cols;
    dst[(size_t)c * rows + r] = (__bf16)src[i];
  }
}

// ---------------------------------------------------------------- QKV GEMM
// X: bf16 [CBT, CD] row-major.  W: bf16 [CH, CD, CHD].  Out: bf16 [(b*CH+h)*CT + t][CHD]
// grid: (CBT/128, CH), block 256 (8 waves); each wave -> 16 rows x 64 cols.
// Double-buffered staging: X tile via one TDM DMA (wave 0), W tile via
// per-lane async-to-LDS. Copy of tile k+1 overlaps WMMAs on tile k.
__global__ __launch_bounds__(256)
void qkv_gemm(const __bf16* __restrict__ X, const __bf16* __restrict__ W,
              __bf16* __restrict__ Out) {
  __shared__ __bf16 Xs[2][128 * 32];
  __shared__ __bf16 Ws[2][32 * 64];

  const int tid   = threadIdx.x;
  const int wave  = tid >> 5;
  const int lane  = tid & 31;
  const int row16 = lane & 15;
  const int hi    = lane >> 4;

  const int h    = blockIdx.y;
  const int row0 = blockIdx.x * 128;
  const __bf16* Wh = W + (size_t)h * CD * CHD;

  const int rw = tid >> 3;          // W-tile row
  const int cw = (tid & 7) * 8;     // W-tile col

  auto stage = [&](int buf, int kk) {
    if (wave == 0) {
      tdm_load_tile_2d(lds_off(&Xs[buf][0]),
                       X + (size_t)row0 * CD + kk,
                       /*tile_w=*/32, /*tile_h=*/128, /*row_stride=*/CD);
    }
    async_ld_b128(lds_off(&Ws[buf][rw * 64 + cw]),
                  Wh + (size_t)(kk + rw) * CHD + cw);
  };

  v8f acc[4] = {};

  stage(0, 0);
  wait_async();
  wait_tensor();
  __syncthreads();

  int buf = 0;
  for (int kk = 0; kk < CD; kk += 32) {
    if (kk + 32 < CD) stage(buf ^ 1, kk + 32);   // async copy of next tile

    // A fragment: rows wave*16 .. +15, K = kk..kk+31 (lane=M, regs=K layout)
    const __bf16* ap = &Xs[buf][(wave * 16 + row16) * 32 + hi * 8];
    v16bf a = ld_frag(ap, ap + 16);
#pragma unroll
    for (int n = 0; n < 4; ++n) {
      const __bf16* bp = &Ws[buf][n * 16 + row16];  // lane=N, elems=K (strided)
      acc[n] = wmma_bf16(a, ld_colk(bp, 64, hi), acc[n]);
    }

    wait_async();
    wait_tensor();
    __syncthreads();
    buf ^= 1;
  }

  // C layout: VGPR j, lanes 0-15 -> M=j, lanes 16-31 -> M=j+8; col = lane&15
#pragma unroll
  for (int n = 0; n < 4; ++n) {
#pragma unroll
    for (int j = 0; j < 8; ++j) {
      int gr = row0 + wave * 16 + j + 8 * hi;
      int bb = gr >> 11;            // / CT
      int t  = gr & (CT - 1);
      Out[((size_t)(bb * CH + h) * CT + t) * CHD + n * 16 + row16] = (__bf16)acc[n][j];
    }
  }
}

// ---------------------------------------------------------------- flash attention
// Q/K/V: bf16 [(b*CH+h)*CT + t][CHD].  O: bf16 [b*CT + t][CD] (col = h*64 + k)
// grid: (CT/128, CH, CB), block 256; each wave owns 16 q-rows x 64 hd.
__global__ __launch_bounds__(256)
void attn(const __bf16* __restrict__ Q, const __bf16* __restrict__ Km,
          const __bf16* __restrict__ Vm, __bf16* __restrict__ O) {
  __shared__ __bf16 Kts[64 * 32];     // transposed: [hd][s]
  __shared__ __bf16 Vs[32 * 64];      // [s][hd]
  __shared__ __bf16 Ps[8 * 16 * 32];  // per-wave P scratch [wave][M][s]

  const int tid   = threadIdx.x;
  const int wave  = tid >> 5;
  const int lane  = tid & 31;
  const int row16 = lane & 15;
  const int hi    = lane >> 4;

  const int h  = blockIdx.y;
  const int b  = blockIdx.z;
  const int q0 = blockIdx.x * 128;
  const int qw = q0 + wave * 16;

  const size_t headoff = (size_t)(b * CH + h) * CT * CHD;
  const __bf16* Qh = Q  + headoff;
  const __bf16* Kh = Km + headoff;
  const __bf16* Vh = Vm + headoff;

  // Q A-fragments (held for the whole pass): K slabs 0..31 and 32..63
  const __bf16* qrow = Qh + (size_t)(qw + row16) * CHD;
  v16bf qa0 = ld_frag(qrow + hi * 8,      qrow + hi * 8 + 16);
  v16bf qa1 = ld_frag(qrow + 32 + hi * 8, qrow + 32 + hi * 8 + 16);

  v8f o0 = {}, o1 = {}, o2 = {}, o3 = {};
  float m[8], l[8];
#pragma unroll
  for (int j = 0; j < 8; ++j) { m[j] = -1e30f; l[j] = 0.0f; }

  const float scale = 0.125f;        // 1/sqrt(64)
  const int smax = q0 + 127;         // causal upper bound for this block
  __bf16* Psw = &Ps[wave * 512];

  for (int s0 = 0; s0 <= smax; s0 += 32) {
    {
      int r = tid >> 3;
      int c = (tid & 7) * 8;
      // K staged transposed for contiguous B-fragments (manual: needs scatter)
      v8bf kv = *(const v8bf*)(Kh + (size_t)(s0 + r) * CHD + c);
#pragma unroll
      for (int i = 0; i < 8; ++i) Kts[(c + i) * 32 + r] = kv[i];
      // V tile (32x64 contiguous) via one Tensor Data Mover DMA
      if (wave == 0) {
        tdm_load_tile_2d(lds_off(&Vs[0]), Vh + (size_t)s0 * CHD,
                         /*tile_w=*/64, /*tile_h=*/32, /*row_stride=*/CHD);
      }
      __builtin_prefetch(Kh + (size_t)(s0 + 32 + r) * CHD + c, 0, 0);
    }
    wait_tensor();
    __syncthreads();

    // S = Q K^T for two 16-column halves (contraction hd=64 -> 2 WMMAs each)
    v8f s_c[2];
#pragma unroll
    for (int f = 0; f < 2; ++f) {
      const __bf16* k0 = &Kts[f * 16 + row16];
      const __bf16* k1 = &Kts[32 * 32 + f * 16 + row16];
      v8f s = {};
      s = wmma_bf16(qa0, ld_colk(k0, 32, hi), s);
      s = wmma_bf16(qa1, ld_colk(k1, 32, hi), s);
      s_c[f] = s;
    }

    // online softmax over the 32 new columns
#pragma unroll
    for (int j = 0; j < 8; ++j) {
      int qr = qw + j + 8 * hi;
      float v0 = s_c[0][j] * scale;
      float v1 = s_c[1][j] * scale;
      if (s0 + row16      > qr) v0 = -1e30f;
      if (s0 + 16 + row16 > qr) v1 = -1e30f;
      float pm = fmaxf(v0, v1);
#pragma unroll
      for (int off = 1; off < 16; off <<= 1) pm = fmaxf(pm, __shfl_xor(pm, off, 32));
      float mn = fmaxf(m[j], pm);
      float p0 = __expf(v0 - mn);
      float p1 = __expf(v1 - mn);
      float alpha = __expf(m[j] - mn);
      m[j] = mn;
      float rs = p0 + p1;
#pragma unroll
      for (int off = 1; off < 16; off <<= 1) rs += __shfl_xor(rs, off, 32);
      l[j] = l[j] * alpha + rs;
      o0[j] *= alpha; o1[j] *= alpha; o2[j] *= alpha; o3[j] *= alpha;
      // spill P into per-wave LDS in [M][s] order for the A-fragment reload
      int M = j + 8 * hi;
      Psw[M * 32 + row16]      = (__bf16)p0;
      Psw[M * 32 + 16 + row16] = (__bf16)p1;
    }
    __syncthreads();   // make Ps visible + order LDS within the wave

    // O += P(16x32) * V(32x64)
    {
      const __bf16* pp = &Psw[row16 * 32 + hi * 8];
      v16bf pa = ld_frag(pp, pp + 16);
      const __bf16* vb;
      vb = &Vs[ 0 + row16]; o0 = wmma_bf16(pa, ld_colk(vb, 64, hi), o0);
      vb = &Vs[16 + row16]; o1 = wmma_bf16(pa, ld_colk(vb, 64, hi), o1);
      vb = &Vs[32 + row16]; o2 = wmma_bf16(pa, ld_colk(vb, 64, hi), o2);
      vb = &Vs[48 + row16]; o3 = wmma_bf16(pa, ld_colk(vb, 64, hi), o3);
    }
    __syncthreads();   // protect Kts/Vs/Ps before next stage
  }

  // normalize and write concat-head output (bf16 for the final GEMM)
#pragma unroll
  for (int j = 0; j < 8; ++j) {
    float inv = 1.0f / l[j];
    int t = qw + j + 8 * hi;
    __bf16* op = O + ((size_t)(b * CT + t)) * CD + h * CHD;
    op[ 0 + row16] = (__bf16)(o0[j] * inv);
    op[16 + row16] = (__bf16)(o1[j] * inv);
    op[32 + row16] = (__bf16)(o2[j] * inv);
    op[48 + row16] = (__bf16)(o3[j] * inv);
  }
}

// ---------------------------------------------------------------- output projection
// X: bf16 [CBT, CD] (attention output). WT: bf16 [CD, CD] = Wo^T (WT[d][j]).
// out[i][j] = sum_d X[i][d]*WT[d][j] + bias[j], f32.
// grid: (CBT/128, CD/64), block 256. Double-buffered TDM + async staging.
__global__ __launch_bounds__(256)
void out_gemm(const __bf16* __restrict__ X, const __bf16* __restrict__ WT,
              const float* __restrict__ bias, float* __restrict__ out) {
  __shared__ __bf16 Xs[2][128 * 32];
  __shared__ __bf16 Ws[2][32 * 64];

  const int tid   = threadIdx.x;
  const int wave  = tid >> 5;
  const int lane  = tid & 31;
  const int row16 = lane & 15;
  const int hi    = lane >> 4;

  const int row0 = blockIdx.x * 128;
  const int nb   = blockIdx.y * 64;

  const int rw = tid >> 3;
  const int cw = (tid & 7) * 8;

  auto stage = [&](int buf, int kk) {
    if (wave == 0) {
      tdm_load_tile_2d(lds_off(&Xs[buf][0]),
                       X + (size_t)row0 * CD + kk,
                       /*tile_w=*/32, /*tile_h=*/128, /*row_stride=*/CD);
    }
    async_ld_b128(lds_off(&Ws[buf][rw * 64 + cw]),
                  WT + (size_t)(kk + rw) * CD + nb + cw);
  };

  v8f acc[4] = {};

  stage(0, 0);
  wait_async();
  wait_tensor();
  __syncthreads();

  int buf = 0;
  for (int kk = 0; kk < CD; kk += 32) {
    if (kk + 32 < CD) stage(buf ^ 1, kk + 32);

    const __bf16* ap = &Xs[buf][(wave * 16 + row16) * 32 + hi * 8];
    v16bf a = ld_frag(ap, ap + 16);
#pragma unroll
    for (int n = 0; n < 4; ++n) {
      const __bf16* bp = &Ws[buf][n * 16 + row16];
      acc[n] = wmma_bf16(a, ld_colk(bp, 64, hi), acc[n]);
    }

    wait_async();
    wait_tensor();
    __syncthreads();
    buf ^= 1;
  }

#pragma unroll
  for (int n = 0; n < 4; ++n) {
#pragma unroll
    for (int j = 0; j < 8; ++j) {
      int gr  = row0 + wave * 16 + j + 8 * hi;
      int col = nb + n * 16 + row16;
      out[(size_t)gr * CD + col] = acc[n][j] + bias[col];
    }
  }
}

// ---------------------------------------------------------------- launch
extern "C" void kernel_launch(void* const* d_in, const int* in_sizes, int n_in,
                              void* d_out, int out_size, void* d_ws, size_t ws_size,
                              hipStream_t stream) {
  (void)in_sizes; (void)n_in; (void)out_size; (void)ws_size;

  const float* x  = (const float*)d_in[0];
  const float* Wq = (const float*)d_in[1];
  const float* Wk = (const float*)d_in[2];
  const float* Wv = (const float*)d_in[3];
  const float* Wo = (const float*)d_in[4];
  const float* bo = (const float*)d_in[5];

  char* ws = (char*)d_ws;
  size_t off = 0;
  __bf16* xb   = (__bf16*)(ws + off); off += (size_t)CBT * CD * 2;        // 16 MB
  __bf16* wqb  = (__bf16*)(ws + off); off += (size_t)CH * CD * CHD * 2;   //  2 MB
  __bf16* wkb  = (__bf16*)(ws + off); off += (size_t)CH * CD * CHD * 2;
  __bf16* wvb  = (__bf16*)(ws + off); off += (size_t)CH * CD * CHD * 2;
  __bf16* wotb = (__bf16*)(ws + off); off += (size_t)CD * CD * 2;         //  2 MB
  __bf16* Qb   = (__bf16*)(ws + off); off += (size_t)CB * CH * CT * CHD * 2; // 16 MB
  __bf16* Kb   = (__bf16*)(ws + off); off += (size_t)CB * CH * CT * CHD * 2;
  __bf16* Vb   = (__bf16*)(ws + off); off += (size_t)CB * CH * CT * CHD * 2;
  __bf16* Ob   = (__bf16*)(ws + off); off += (size_t)CBT * CD * 2;        // 16 MB

  const int nx = CBT * CD;
  const int nw = CH * CD * CHD;
  cvt_bf16<<<(nx + 255) / 256, 256, 0, stream>>>(x,  xb,  nx);
  cvt_bf16<<<(nw + 255) / 256, 256, 0, stream>>>(Wq, wqb, nw);
  cvt_bf16<<<(nw + 255) / 256, 256, 0, stream>>>(Wk, wkb, nw);
  cvt_bf16<<<(nw + 255) / 256, 256, 0, stream>>>(Wv, wvb, nw);
  cvt_bf16_T<<<(CD * CD + 255) / 256, 256, 0, stream>>>(Wo, wotb, CD, CD);

  dim3 gq(CBT / 128, CH);
  qkv_gemm<<<gq, 256, 0, stream>>>(xb, wqb, Qb);
  qkv_gemm<<<gq, 256, 0, stream>>>(xb, wkb, Kb);
  qkv_gemm<<<gq, 256, 0, stream>>>(xb, wvb, Vb);

  attn<<<dim3(CT / 128, CH, CB), 256, 0, stream>>>(Qb, Kb, Vb, Ob);

  out_gemm<<<dim3(CBT / 128, CD / 64), 256, 0, stream>>>(Ob, wotb, bo, (float*)d_out);
}
